// SelfAttention_3058016714914
// MI455X (gfx1250) — matrix operand order repaired and checked by reading, not compiled, not run on hardware
//
#include <hip/hip_runtime.h>
#include <hip/hip_bf16.h>

// ---------------------------------------------------------------------------
// MI455X (gfx1250) self-attention, bf16 WMMA pipeline, f32 accumulate.
// Round 3: sched_group_barrier (VMEM-read then WMMA groups) to force
// load/WMMA overlap -> partial s_wait_loadcnt instead of wait-0 per wmma.
// ---------------------------------------------------------------------------

typedef __attribute__((ext_vector_type(16))) __bf16 v16bf;
typedef __attribute__((ext_vector_type(8)))  __bf16 v8bf;
typedef __attribute__((ext_vector_type(8)))  float  v8f;

constexpr int EMBED = 256;
constexpr int HEADS = 8;
constexpr int NB    = 8;     // batch
constexpr int T     = 1024;
constexpr int HE    = HEADS * EMBED;   // 2048

// workspace layout (bf16 element offsets)
constexpr size_t SZ_XB  = (size_t)NB * T * EMBED;          // 2,097,152
constexpr size_t SZ_W   = (size_t)HEADS * EMBED * EMBED;   // 524,288
constexpr size_t SZ_WO  = (size_t)EMBED * HE;              // 524,288
constexpr size_t SZ_QKV = (size_t)HEADS * NB * T * EMBED;  // 16,777,216
constexpr size_t OFF_XB = 0;
constexpr size_t OFF_WQ = OFF_XB + SZ_XB;
constexpr size_t OFF_WK = OFF_WQ + SZ_W;
constexpr size_t OFF_WV = OFF_WK + SZ_W;
constexpr size_t OFF_WO = OFF_WV + SZ_W;
constexpr size_t OFF_Q  = OFF_WO + SZ_WO;
constexpr size_t OFF_K  = OFF_Q  + SZ_QKV;
constexpr size_t OFF_VT = OFF_K  + SZ_QKV;   // V stored transposed [h,b,d,t]
constexpr size_t OFF_O  = OFF_VT + SZ_QKV;   // concat heads [b,t,h*e]

// sched_group_barrier masks
#define SG_WMMA   0x008
#define SG_VMREAD 0x020

__device__ inline v8f wmma_bf16(v16bf a, v16bf b, v8f c) {
  return __builtin_amdgcn_wmma_f32_16x16x32_bf16(
      /*neg_a=*/false, a, /*neg_b=*/false, b,
      /*c_mod=*/(short)0, c, /*reuse_a=*/false, /*reuse_b=*/false);
}

// A-operand (16xK row tile, 16-bit): lane(0-15)=row m, kh=lane>>4.
// elems 0..15 -> K = kbase + kh*16 + i
__device__ inline v16bf load_a16(const __bf16* rowp, int kbase, int kh) {
  v8bf lo = *(const v8bf*)(rowp + kbase + kh * 16);
  v8bf hi = *(const v8bf*)(rowp + kbase + kh * 16 + 8);
  v16bf a;
#pragma unroll
  for (int i = 0; i < 8; ++i) { a[i] = lo[i]; a[8 + i] = hi[i]; }
  return a;
}

// B-operand (Kx16, 16-bit): lane holds column n=lane&15; elems i -> K = kh*16 + i
__device__ inline v16bf load_b16(const __bf16* p) {
  return *(const v16bf*)(p);
}

// ---------------------------------------------------------------------------
__global__ void __launch_bounds__(256)
cvt_f32_bf16(const float* __restrict__ src, __bf16* __restrict__ dst, int n) {
  int i = blockIdx.x * blockDim.x + threadIdx.x;
  int stride = gridDim.x * blockDim.x;
  for (; i < n; i += stride) dst[i] = (__bf16)src[i];
}

// ---------------------------------------------------------------------------
// One wave -> one 16(t) x 256(d) tile of q, k, or v for one (proj, head, batch).
// q,k stored row-major [h,b,t,d]; v stored transposed [h,b,d,t].
__global__ void __launch_bounds__(256)
qkv_proj(const __bf16* __restrict__ xb,
         const __bf16* __restrict__ wq, const __bf16* __restrict__ wk,
         const __bf16* __restrict__ wv,
         __bf16* __restrict__ qb, __bf16* __restrict__ kb,
         __bf16* __restrict__ vtb) {
  const int lane = threadIdx.x & 31;
  const int wid  = blockIdx.x * (blockDim.x >> 5) + (threadIdx.x >> 5);
  const int tt = wid & 63;  int tmp = wid >> 6;
  const int b  = tmp & 7;   tmp >>= 3;
  const int h  = tmp & 7;   const int p = tmp >> 3;  // 0=q 1=k 2=v
  const int n  = lane & 15, kh = lane >> 4;
  const int t0 = tt * 16;
  const size_t hb = (size_t)h * NB + b;

  const __bf16* W = (p == 0 ? wq : (p == 1 ? wk : wv)) + (size_t)h * EMBED * EMBED;

  // preload A (x rows) for all 8 K-chunks, reused across 16 N-tiles
  v16bf a[8];
  const __bf16* xrow = xb + ((size_t)b * T + t0 + n) * EMBED;
#pragma unroll
  for (int kc = 0; kc < 8; ++kc) a[kc] = load_a16(xrow, kc * 32, kh);

#pragma unroll 1
  for (int nt = 0; nt < 16; ++nt) {
    const __bf16* wrow = W + (size_t)(nt * 16 + n) * EMBED;  // y = x @ W^T
    v16bf bb[8];
#pragma unroll
    for (int kc = 0; kc < 8; ++kc) bb[kc] = load_b16(wrow + kc * 32 + kh * 16);
    v8f c = {};
#pragma unroll
    for (int kc = 0; kc < 8; ++kc) c = wmma_bf16(a[kc], bb[kc], c);

    // schedule: all 16 b128 reads first, then the 8-WMMA chain
    __builtin_amdgcn_sched_group_barrier(SG_VMREAD, 16, 0);
    __builtin_amdgcn_sched_group_barrier(SG_WMMA,    8, 0);

    if (p == 2) {
#pragma unroll
      for (int r = 0; r < 8; ++r) {
        int trow = t0 + r + 8 * kh;
        vtb[(hb * EMBED + nt * 16 + n) * T + trow] = (__bf16)c[r];
      }
    } else {
      __bf16* dst = (p == 0 ? qb : kb);
#pragma unroll
      for (int r = 0; r < 8; ++r) {
        int trow = t0 + r + 8 * kh;
        dst[(hb * T + trow) * EMBED + nt * 16 + n] = (__bf16)c[r];
      }
    }
  }
}

// ---------------------------------------------------------------------------
// Flash attention: one wave per (head, batch, 16-row Q tile).
__global__ void __launch_bounds__(256)
flash_attn(const __bf16* __restrict__ qb, const __bf16* __restrict__ kb,
           const __bf16* __restrict__ vtb, __bf16* __restrict__ ob) {
  __shared__ __attribute__((aligned(16))) __bf16 plds[8][16][32];
  const int lane = threadIdx.x & 31;
  const int w    = threadIdx.x >> 5;
  const int wid  = blockIdx.x * 8 + w;
  const int it = wid & 63;  int tmp = wid >> 6;
  const int b = tmp & 7;    const int h = tmp >> 3;
  const int n = lane & 15, kh = lane >> 4;
  const int i0 = it * 16;
  const size_t hb = (size_t)h * NB + b;

  v16bf qa[8];
  const __bf16* qrow = qb + (hb * T + i0 + n) * EMBED;
#pragma unroll
  for (int kc = 0; kc < 8; ++kc) qa[kc] = load_a16(qrow, kc * 32, kh);

  v8f o[16];
#pragma unroll
  for (int nt = 0; nt < 16; ++nt) o[nt] = {};
  float mrow[8], lrow[8];
#pragma unroll
  for (int r = 0; r < 8; ++r) { mrow[r] = -__builtin_inff(); lrow[r] = 0.f; }

  const int nchunk = (i0 + 16 + 31) >> 5;   // cover causal region in 32-j steps
  for (int jc = 0; jc < nchunk; ++jc) {
    const int j0 = jc * 32;

    // ---- load all K B-tiles for both 16-j halves up front --------------
    v16bf kbt[2][8];
#pragma unroll
    for (int jh = 0; jh < 2; ++jh) {
      const __bf16* krow = kb + (hb * T + j0 + jh * 16 + n) * EMBED;
#pragma unroll
      for (int kc = 0; kc < 8; ++kc)
        kbt[jh][kc] = load_b16(krow + kc * 32 + kh * 16);
    }

    // ---- S = Q @ K^T : two 16x16 tiles ---------------------------------
    v8f s[2];
#pragma unroll
    for (int jh = 0; jh < 2; ++jh) {
      v8f c = {};
#pragma unroll
      for (int kc = 0; kc < 8; ++kc) c = wmma_bf16(qa[kc], kbt[jh][kc], c);
      const int j = j0 + jh * 16 + n;
#pragma unroll
      for (int r = 0; r < 8; ++r) {
        int i = i0 + r + 8 * kh;
        if (j > i) c[r] = -__builtin_inff();   // causal mask
      }
      s[jh] = c;
    }

    // ---- online softmax (C layout: VGPR r = row r+8*kh) ----------------
    float rmax[8];
#pragma unroll
    for (int r = 0; r < 8; ++r) rmax[r] = fmaxf(s[0][r], s[1][r]);
#pragma unroll
    for (int mk = 1; mk < 16; mk <<= 1)
#pragma unroll
      for (int r = 0; r < 8; ++r)
        rmax[r] = fmaxf(rmax[r], __shfl_xor(rmax[r], mk, 32));

    float mnew[8], alpha[8];
#pragma unroll
    for (int r = 0; r < 8; ++r) {
      mnew[r]  = fmaxf(mrow[r], rmax[r]);
      alpha[r] = __expf(mrow[r] - mnew[r]);
      mrow[r]  = mnew[r];
    }

    float rsum[8];
#pragma unroll
    for (int r = 0; r < 8; ++r) rsum[r] = 0.f;
#pragma unroll
    for (int jh = 0; jh < 2; ++jh)
#pragma unroll
      for (int r = 0; r < 8; ++r) {
        float pv = __expf(s[jh][r] - mnew[r]);
        rsum[r] += pv;
        plds[w][r + 8 * kh][jh * 16 + n] = (__bf16)pv;
      }
#pragma unroll
    for (int mk = 1; mk < 16; mk <<= 1)
#pragma unroll
      for (int r = 0; r < 8; ++r) rsum[r] += __shfl_xor(rsum[r], mk, 32);
#pragma unroll
    for (int r = 0; r < 8; ++r) lrow[r] = lrow[r] * alpha[r] + rsum[r];

    // rescale O accumulators
#pragma unroll
    for (int nt = 0; nt < 16; ++nt)
#pragma unroll
      for (int r = 0; r < 8; ++r) o[nt][r] *= alpha[r];

    // ---- V B-tiles (independent of softmax -> scheduler hoists them) ---
    v16bf vb[2][8];
#pragma unroll
    for (int g = 0; g < 2; ++g)
#pragma unroll
      for (int q = 0; q < 8; ++q) {
        const __bf16* vrow = vtb + (hb * EMBED + (g * 8 + q) * 16 + n) * T + j0;
        vb[g][q] = load_b16(vrow + kh * 16);
      }

    // re-layout P: LDS round trip C-layout -> A-operand layout
    // (same-wave LDS ops are processed in order; compiler inserts dscnt wait)
    v16bf pa = load_a16(&plds[w][n][0], 0, kh);

    // ---- O += P @ V ----------------------------------------------------
#pragma unroll
    for (int g = 0; g < 2; ++g)
#pragma unroll
      for (int q = 0; q < 8; ++q)
        o[g * 8 + q] = wmma_bf16(pa, vb[g][q], o[g * 8 + q]);

    // schedule: K loads | QK wmmas | V loads (under softmax) | PV wmmas
    __builtin_amdgcn_sched_group_barrier(SG_VMREAD, 32, 0);
    __builtin_amdgcn_sched_group_barrier(SG_WMMA,   16, 0);
    __builtin_amdgcn_sched_group_barrier(SG_VMREAD, 32, 0);
    __builtin_amdgcn_sched_group_barrier(SG_WMMA,   16, 0);
  }

  // normalize and store concat-head layout [b, t, h*e]
  float rl[8];
#pragma unroll
  for (int r = 0; r < 8; ++r) rl[r] = __builtin_amdgcn_rcpf(lrow[r]);
#pragma unroll
  for (int nt = 0; nt < 16; ++nt)
#pragma unroll
    for (int r = 0; r < 8; ++r) {
      int t = i0 + r + 8 * kh;
      ob[((size_t)b * T + t) * HE + h * EMBED + nt * 16 + n] =
          (__bf16)(o[nt][r] * rl[r]);
    }
}

// ---------------------------------------------------------------------------
// Output projection: y = O[b,t,2048] @ Wo^T + bo  -> f32 [b,t,256]
__global__ void __launch_bounds__(256)
out_proj(const __bf16* __restrict__ ob, const __bf16* __restrict__ wob,
         const float* __restrict__ bo, float* __restrict__ out) {
  const int lane = threadIdx.x & 31;
  const int wid  = blockIdx.x * 8 + (threadIdx.x >> 5);
  const int tt = wid & 63;  const int b = wid >> 6;
  const int n = lane & 15, kh = lane >> 4;
  const int t0 = tt * 16;

  v8f c[16];
#pragma unroll
  for (int nt = 0; nt < 16; ++nt) c[nt] = {};

  const __bf16* arow = ob + ((size_t)b * T + t0 + n) * HE;
#pragma unroll 1
  for (int kc = 0; kc < 64; ++kc) {
    v16bf a = load_a16(arow, kc * 32, kh);
    v16bf bb[16];
#pragma unroll
    for (int q = 0; q < 16; ++q) {
      const __bf16* wrow = wob + (size_t)(q * 16 + n) * HE + kc * 32 + kh * 16;
      bb[q] = load_b16(wrow);
    }
#pragma unroll
    for (int q = 0; q < 16; ++q) c[q] = wmma_bf16(a, bb[q], c[q]);

    // schedule: A (2 reads) + 16 W tiles (32 reads) first, then 16 WMMAs
    __builtin_amdgcn_sched_group_barrier(SG_VMREAD, 34, 0);
    __builtin_amdgcn_sched_group_barrier(SG_WMMA,   16, 0);
  }
#pragma unroll
  for (int nt = 0; nt < 16; ++nt) {
    float bias = bo[nt * 16 + n];
#pragma unroll
    for (int r = 0; r < 8; ++r) {
      int t = t0 + r + 8 * kh;
      out[((size_t)b * T + t) * EMBED + nt * 16 + n] = c[nt][r] + bias;
    }
  }
}

// ---------------------------------------------------------------------------
extern "C" void kernel_launch(void* const* d_in, const int* in_sizes, int n_in,
                              void* d_out, int out_size, void* d_ws, size_t ws_size,
                              hipStream_t stream) {
  const float* x  = (const float*)d_in[0];
  const float* Wq = (const float*)d_in[1];
  const float* Wk = (const float*)d_in[2];
  const float* Wv = (const float*)d_in[3];
  const float* Wo = (const float*)d_in[4];
  const float* bo = (const float*)d_in[5];
  float* out = (float*)d_out;

  __bf16* ws  = (__bf16*)d_ws;
  __bf16* xb  = ws + OFF_XB;
  __bf16* wqb = ws + OFF_WQ;
  __bf16* wkb = ws + OFF_WK;
  __bf16* wvb = ws + OFF_WV;
  __bf16* wob = ws + OFF_WO;
  __bf16* qb  = ws + OFF_Q;
  __bf16* kb  = ws + OFF_K;
  __bf16* vtb = ws + OFF_VT;
  __bf16* ob  = ws + OFF_O;

  cvt_f32_bf16<<<2048, 256, 0, stream>>>(x,  xb,  (int)SZ_XB);
  cvt_f32_bf16<<<1024, 256, 0, stream>>>(Wq, wqb, (int)SZ_W);
  cvt_f32_bf16<<<1024, 256, 0, stream>>>(Wk, wkb, (int)SZ_W);
  cvt_f32_bf16<<<1024, 256, 0, stream>>>(Wv, wvb, (int)SZ_W);
  cvt_f32_bf16<<<1024, 256, 0, stream>>>(Wo, wob, (int)SZ_WO);

  // 3 proj * 8 heads * 8 batch * 64 t-tiles = 12288 waves, 8 waves/block
  qkv_proj<<<1536, 256, 0, stream>>>(xb, wqb, wkb, wvb, qb, kb, vtb);
  // 8 heads * 8 batch * 64 t-tiles = 4096 waves
  flash_attn<<<512, 256, 0, stream>>>(qb, kb, vtb, ob);
  // 8 batch * 64 t-tiles = 512 waves
  out_proj<<<64, 256, 0, stream>>>(ob, wob, bo, out);
}
